// SAGNet_12214886990511
// MI455X (gfx1250) — matrix operand order repaired and checked by reading, not compiled, not run on hardware
//
#include <hip/hip_runtime.h>
#include <hip/hip_bf16.h>
#include <math.h>

// ---------------------------------------------------------------------------
// Problem constants (match reference)
// ---------------------------------------------------------------------------
#define NNODE 50000
#define NEDGE 600000
#define FDIM 128
#define NGRAPH 100
#define NCLS 10

typedef __attribute__((ext_vector_type(16))) __bf16 v16bf;
typedef __attribute__((ext_vector_type(8)))  __bf16 v8bf;
typedef __attribute__((ext_vector_type(8)))  float  v8f;
typedef int v4i __attribute__((vector_size(16)));
typedef __attribute__((address_space(1))) v4i gv4i;   // global int4
typedef __attribute__((address_space(3))) v4i lv4i;   // LDS int4

#if __has_builtin(__builtin_amdgcn_global_load_async_to_lds_b128)
#define HAVE_ASYNC_LDS 1
#else
#define HAVE_ASYNC_LDS 0
#endif

// ---------------------------------------------------------------------------
// Utility fills
// ---------------------------------------------------------------------------
__global__ void fill_f32(float* __restrict__ p, float v, int n) {
  int i = blockIdx.x * blockDim.x + threadIdx.x;
  if (i < n) p[i] = v;
}
__global__ void fill_i32(int* __restrict__ p, int v, int n) {
  int i = blockIdx.x * blockDim.x + threadIdx.x;
  if (i < n) p[i] = v;
}

// ---------------------------------------------------------------------------
// f32 -> bf16 convert (GEMM A operand)
// ---------------------------------------------------------------------------
__global__ void conv_bf16(const float* __restrict__ src, __bf16* __restrict__ dst, int n) {
  int i = blockIdx.x * blockDim.x + threadIdx.x;
  if (i < n) dst[i] = (__bf16)src[i];
}

// ---------------------------------------------------------------------------
// Pack 128x128 f32 weight (row-major, K x N) into WMMA-native bf16 B fragments.
// Layout: [nt (8)][kt (4)][lane (32)][elem (16)] bf16.
// B fragment (contiguous k order, same as A): lane l holds col n = nt*16 + (l&15);
//   elem i -> k = kt*32 + (l>>4)*16 + i.
// ---------------------------------------------------------------------------
__global__ void pack_w(const float* __restrict__ W, __bf16* __restrict__ Bp) {
  int idx = blockIdx.x * blockDim.x + threadIdx.x;       // 8*4*32*16 = 16384
  if (idx >= 8 * 4 * 32 * 16) return;
  int i    = idx & 15;
  int lane = (idx >> 4) & 31;
  int kt   = (idx >> 9) & 3;
  int nt   = (idx >> 11) & 7;
  int n = nt * 16 + (lane & 15);
  int k = kt * 32 + ((lane >> 4) * 16) + i;
  Bp[idx] = (__bf16)W[k * FDIM + n];
}

// ---------------------------------------------------------------------------
// WMMA GEMM: C(nrows x 128) = A(nrows x 128, bf16) * Bpacked(128x128, bf16)
// The packed weight tile (32 KB) is staged into LDS once per workgroup via
// async global->LDS (ASYNCcnt / s_wait_asynccnt), then each wave computes a
// 16-row tile across all 8 column tiles: 4 x 8 wmma_f32_16x16x32_bf16.
// ---------------------------------------------------------------------------
__global__ __launch_bounds__(128) void gemm_wmma(
    const __bf16* __restrict__ A, const __bf16* __restrict__ Bp,
    float* __restrict__ C, int nrows) {
  __shared__ __align__(32) __bf16 sB[8 * 4 * 32 * 16];   // 32 KB packed weight

  int tid = threadIdx.x;
#if HAVE_ASYNC_LDS
  {
    // 2048 x 16-byte chunks, 16 per thread, fully async into LDS
#pragma unroll
    for (int c = 0; c < 16; ++c) {
      unsigned byteoff = (unsigned)((c * 128 + tid) * 16);
      __builtin_amdgcn_global_load_async_to_lds_b128(
          (gv4i*)((const char*)Bp + byteoff),
          (lv4i*)((char*)sB + byteoff),
          0, 0);
    }
#if __has_builtin(__builtin_amdgcn_s_wait_asynccnt)
    __builtin_amdgcn_s_wait_asynccnt(0);
#else
    asm volatile("s_wait_asynccnt 0" ::: "memory");
#endif
  }
#else
  {
    const int4* gsrc = (const int4*)Bp;
    int4* ldst = (int4*)sB;
    for (int c = tid; c < 2048; c += 128) ldst[c] = gsrc[c];
  }
#endif
  __syncthreads();

  int wave = blockIdx.x * (blockDim.x >> 5) + (threadIdx.x >> 5);
  int lane = threadIdx.x & 31;
  int m0 = wave * 16;
  if (m0 >= nrows) return;                 // wave-uniform: EXEC stays all-1s

  int mrow  = m0 + (lane & 15);
  int khalf = lane >> 4;                   // 0 or 1

  v8f acc[8] = {};                         // 8 output tiles of 16x16 f32

  for (int kt = 0; kt < 4; ++kt) {
    int kk = kt * 32;
    // A fragment: two contiguous groups of 8 bf16 per lane
    //   elem i   -> k = kk + khalf*16 + i        (i=0..7)
    //   elem 8+i -> k = kk + khalf*16 + 8 + i
    const __bf16* arow = A + (size_t)mrow * FDIM + kk + khalf * 16;
    v8bf g0 = *(const v8bf*)(arow);
    v8bf g1 = *(const v8bf*)(arow + 8);
    v16bf afrag;
#pragma unroll
    for (int i = 0; i < 8; ++i) { afrag[i] = g0[i]; afrag[8 + i] = g1[i]; }

#pragma unroll
    for (int nt = 0; nt < 8; ++nt) {
      v16bf bfrag = *(const v16bf*)(sB + (((nt * 4 + kt) * 32 + lane) << 4));
      acc[nt] = __builtin_amdgcn_wmma_f32_16x16x32_bf16(
          false, afrag, false, bfrag, (short)0, acc[nt], false, false);
    }
  }

  // C layout: VGPR j -> row m0 + j + khalf*8, col = nt*16 + (lane&15)
  int mbase = m0 + khalf * 8;
  int ncol  = lane & 15;
#pragma unroll
  for (int nt = 0; nt < 8; ++nt) {
#pragma unroll
    for (int j = 0; j < 8; ++j) {
      C[(size_t)(mbase + j) * FDIM + nt * 16 + ncol] = acc[nt][j];
    }
  }
}

// ---------------------------------------------------------------------------
// Graph bookkeeping (batch is sorted)
// ---------------------------------------------------------------------------
__global__ void count_batch(const int* __restrict__ batch, int* __restrict__ cnt, int n) {
  int i = blockIdx.x * blockDim.x + threadIdx.x;
  if (i < n) atomicAdd(&cnt[batch[i]], 1);
}
__global__ void scan_starts(const int* __restrict__ cnt, int* __restrict__ starts) {
  if (threadIdx.x == 0 && blockIdx.x == 0) {
    int s = 0;
    for (int g = 0; g < NGRAPH; ++g) { starts[g] = s; s += cnt[g]; }
  }
}

// ---------------------------------------------------------------------------
// Edge pipeline
// ---------------------------------------------------------------------------
__global__ void edge_deg(const int* __restrict__ src, const int* __restrict__ dst,
                         const float* __restrict__ mask, float* __restrict__ deg, int nE) {
  int e = blockIdx.x * blockDim.x + threadIdx.x;
  if (e >= nE) return;
  float ew = mask[src[e]] * mask[dst[e]];
  if (ew != 0.f) atomicAdd(&deg[dst[e]], ew);
}
__global__ void calc_norm(const float* __restrict__ deg, float* __restrict__ norm, int n) {
  int i = blockIdx.x * blockDim.x + threadIdx.x;
  if (i < n) norm[i] = rsqrtf(deg[i]);
}
__global__ void calc_coef(const int* __restrict__ src, const int* __restrict__ dst,
                          const float* __restrict__ mask, const float* __restrict__ norm,
                          float* __restrict__ coef, int nE) {
  int e = blockIdx.x * blockDim.x + threadIdx.x;
  if (e >= nE) return;
  int s = src[e], d = dst[e];
  coef[e] = norm[s] * norm[d] * mask[s] * mask[d];
}
// One 32-lane wave per edge; each lane scatters 4 features.
__global__ __launch_bounds__(256) void scatter_feat(
    const int* __restrict__ src, const int* __restrict__ dst,
    const float* __restrict__ xw, const float* __restrict__ coef,
    float* __restrict__ agg, int nE) {
  int t = blockIdx.x * blockDim.x + threadIdx.x;
  int e = t >> 5;
  int lane = t & 31;
  if (e >= nE) return;
  float c = coef[e];
  if (c == 0.f) return;
  int s = src[e], d = dst[e];
  const float* xr = xw + (size_t)s * FDIM;
  float* ar = agg + (size_t)d * FDIM;
  __builtin_prefetch(xr + lane * 4, 0, 1);   // global_prefetch_b8
#pragma unroll
  for (int q = 0; q < 4; ++q) {
    int f = lane + q * 32;
    atomicAdd(&ar[f], xr[f] * c);
  }
}
__global__ void combine_relu(const float* __restrict__ agg, const float* __restrict__ xw,
                             const float* __restrict__ norm, const float* __restrict__ b,
                             float* __restrict__ hout, int n) {
  int idx = blockIdx.x * blockDim.x + threadIdx.x;
  if (idx >= n * FDIM) return;
  int node = idx >> 7, f = idx & 127;
  float nm = norm[node];
  float v = agg[idx] + xw[idx] * nm * nm + b[f];
  hout[idx] = fmaxf(v, 0.f);
}

// ---------------------------------------------------------------------------
// Score conv (H -> 1)
// ---------------------------------------------------------------------------
__global__ __launch_bounds__(256) void gemv_score(
    const float* __restrict__ h, const float* __restrict__ Ws,
    float* __restrict__ slin, int nrows) {
  int wave = blockIdx.x * (blockDim.x >> 5) + (threadIdx.x >> 5);
  int lane = threadIdx.x & 31;
  if (wave >= nrows) return;
  float acc = 0.f;
  const float* row = h + (size_t)wave * FDIM;
#pragma unroll
  for (int q = 0; q < 4; ++q) acc += row[lane + q * 32] * Ws[lane + q * 32];
#pragma unroll
  for (int off = 16; off > 0; off >>= 1) acc += __shfl_xor(acc, off, 32);
  if (lane == 0) slin[wave] = acc;
}
__global__ void scatter_score(const int* __restrict__ src, const int* __restrict__ dst,
                              const float* __restrict__ slin, const float* __restrict__ coef,
                              float* __restrict__ score, int nE) {
  int e = blockIdx.x * blockDim.x + threadIdx.x;
  if (e >= nE) return;
  float c = coef[e];
  if (c != 0.f) atomicAdd(&score[dst[e]], slin[src[e]] * c);
}
__global__ void finalize_score(float* __restrict__ score, const float* __restrict__ slin,
                               const float* __restrict__ norm, const float* __restrict__ bs, int n) {
  int i = blockIdx.x * blockDim.x + threadIdx.x;
  if (i >= n) return;
  float nm = norm[i];
  score[i] = score[i] + slin[i] * nm * nm + bs[0];
}

// ---------------------------------------------------------------------------
// Top-k pooling
// ---------------------------------------------------------------------------
__global__ void accum_active(const float* __restrict__ mask, const int* __restrict__ batch,
                             float* __restrict__ active, int n) {
  int i = blockIdx.x * blockDim.x + threadIdx.x;
  if (i < n) atomicAdd(&active[batch[i]], mask[i]);
}
__global__ __launch_bounds__(256) void topk_mask(
    const float* __restrict__ score, const float* __restrict__ mask,
    const int* __restrict__ starts, const int* __restrict__ cnt,
    const float* __restrict__ active, float* __restrict__ nmask) {
  int g = blockIdx.x;
  int s0 = starts[g], c = cnt[g];
  float kf = ceilf(0.5f * active[g]);
  for (int i = threadIdx.x; i < c; i += blockDim.x) {
    int n = s0 + i;
    float mn = mask[n];
    float keep = 0.f;
    if (mn > 0.f) {
      float sn = score[n];
      int rank = 0;
      for (int j = 0; j < c; ++j) {
        int m2 = s0 + j;
        if (mask[m2] > 0.f) {
          float sj = score[m2];
          rank += (sj > sn || (sj == sn && m2 < n)) ? 1 : 0;
        }
      }
      keep = ((float)rank < kf) ? 1.f : 0.f;
    }
    nmask[n] = mn * keep;
  }
}
__global__ void apply_gate(float* __restrict__ h, const float* __restrict__ score,
                           const float* __restrict__ nmask, int n) {
  int idx = blockIdx.x * blockDim.x + threadIdx.x;
  if (idx >= n * FDIM) return;
  int node = idx >> 7;
  h[idx] = h[idx] * tanhf(score[node]) * nmask[node];
}

// ---------------------------------------------------------------------------
// Readout (max || mean), accumulated into z (G x 256)
// ---------------------------------------------------------------------------
__global__ __launch_bounds__(128) void readout(
    const float* __restrict__ h, const float* __restrict__ nmask,
    const int* __restrict__ starts, const int* __restrict__ cnt,
    float* __restrict__ z) {
  int g = blockIdx.x, f = threadIdx.x;
  int s0 = starts[g], c = cnt[g];
  float sum = 0.f, mx = -INFINITY, ca = 0.f;
  for (int i = 0; i < c; ++i) {
    int n = s0 + i;
    float m = nmask[n];
    float v = h[(size_t)n * FDIM + f];
    sum += v * m;
    if (m > 0.f) mx = fmaxf(mx, v);
    ca += m;
  }
  float mean = sum / fmaxf(ca, 1.f);
  if (mx == -INFINITY) mx = 0.f;
  z[g * 256 + f]       += mx;
  z[g * 256 + 128 + f] += mean;
}

// ---------------------------------------------------------------------------
// Small MLP + log-softmax (G = 100 rows)
// ---------------------------------------------------------------------------
__global__ void linear_k(const float* __restrict__ in, const float* __restrict__ W,
                         const float* __restrict__ b, float* __restrict__ out,
                         int rows, int K, int Ncol, int do_relu) {
  int idx = blockIdx.x * blockDim.x + threadIdx.x;
  if (idx >= rows * Ncol) return;
  int r = idx / Ncol, c = idx % Ncol;
  float acc = b[c];
  for (int k = 0; k < K; ++k) acc += in[r * K + k] * W[k * Ncol + c];
  if (do_relu) acc = fmaxf(acc, 0.f);
  out[idx] = acc;
}
__global__ void log_softmax_k(const float* __restrict__ zin, float* __restrict__ out, int rows) {
  int r = blockIdx.x * blockDim.x + threadIdx.x;
  if (r >= rows) return;
  float mx = -INFINITY;
  for (int c = 0; c < NCLS; ++c) mx = fmaxf(mx, zin[r * NCLS + c]);
  float se = 0.f;
  for (int c = 0; c < NCLS; ++c) se += expf(zin[r * NCLS + c] - mx);
  float l = logf(se);
  for (int c = 0; c < NCLS; ++c) out[r * NCLS + c] = zin[r * NCLS + c] - mx - l;
}

// ---------------------------------------------------------------------------
// Host orchestration
// ---------------------------------------------------------------------------
static inline int cdiv(long long a, long long b) { return (int)((a + b - 1) / b); }

extern "C" void kernel_launch(void* const* d_in, const int* in_sizes, int n_in,
                              void* d_out, int out_size, void* d_ws, size_t ws_size,
                              hipStream_t stream) {
  const int N = NNODE, E = NEDGE, G = NGRAPH;

  const float* x     = (const float*)d_in[0];
  const int*   eidx  = (const int*)d_in[1];
  const int*   batch = (const int*)d_in[2];
  const float* Wm[3]  = {(const float*)d_in[3],  (const float*)d_in[7],  (const float*)d_in[11]};
  const float* bm[3]  = {(const float*)d_in[4],  (const float*)d_in[8],  (const float*)d_in[12]};
  const float* Wsm[3] = {(const float*)d_in[5],  (const float*)d_in[9],  (const float*)d_in[13]};
  const float* bsm[3] = {(const float*)d_in[6],  (const float*)d_in[10], (const float*)d_in[14]};
  const float* Wl1 = (const float*)d_in[15]; const float* bl1 = (const float*)d_in[16];
  const float* Wl2 = (const float*)d_in[17]; const float* bl2 = (const float*)d_in[18];
  const float* Wl3 = (const float*)d_in[19]; const float* bl3 = (const float*)d_in[20];

  const int* esrc = eidx;
  const int* edst = eidx + E;

  // --- workspace carve ---
  char* ws = (char*)d_ws;
  size_t off = 0;
  auto carve = [&](size_t bytes) -> char* {
    char* p = ws + off;
    off += (bytes + 255) & ~(size_t)255;
    return p;
  };
  float* maskA = (float*)carve((size_t)N * 4);
  float* maskB = (float*)carve((size_t)N * 4);
  float* deg   = (float*)carve((size_t)N * 4);
  float* norm  = (float*)carve((size_t)N * 4);
  float* slin  = (float*)carve((size_t)N * 4);
  float* score = (float*)carve((size_t)N * 4);
  float* coef  = (float*)carve((size_t)E * 4);
  float* xw    = (float*)carve((size_t)N * FDIM * 4);
  float* agg   = (float*)carve((size_t)N * FDIM * 4);
  float* hbuf  = (float*)carve((size_t)N * FDIM * 4);
  __bf16* hbf  = (__bf16*)carve((size_t)N * FDIM * 2);
  __bf16* wpk  = (__bf16*)carve((size_t)FDIM * FDIM * 2);
  int*   cnt    = (int*)carve((size_t)G * 4);
  int*   starts = (int*)carve((size_t)G * 4);
  float* active = (float*)carve((size_t)G * 4);
  float* z      = (float*)carve((size_t)G * 256 * 4);
  float* z1     = (float*)carve((size_t)G * 128 * 4);
  float* z2     = (float*)carve((size_t)G * 64 * 4);
  float* z3     = (float*)carve((size_t)G * 16 * 4);
  (void)ws_size; (void)n_in; (void)in_sizes; (void)out_size;

  const int T = 256;
  const int NF = N * FDIM;

  // --- graph bookkeeping (once) ---
  fill_i32<<<cdiv(G, T), T, 0, stream>>>(cnt, 0, G);
  count_batch<<<cdiv(N, T), T, 0, stream>>>(batch, cnt, N);
  scan_starts<<<1, 32, 0, stream>>>(cnt, starts);

  fill_f32<<<cdiv(G * 256, T), T, 0, stream>>>(z, 0.f, G * 256);
  fill_f32<<<cdiv(N, T), T, 0, stream>>>(maskA, 1.f, N);

  const float* curh = x;
  float* maskCur = maskA;
  float* maskNext = maskB;

  for (int layer = 0; layer < 3; ++layer) {
    // xw = curh @ W  (bf16 WMMA, f32 accumulate)
    conv_bf16<<<cdiv(NF, T), T, 0, stream>>>(curh, hbf, NF);
    pack_w<<<cdiv(16384, T), T, 0, stream>>>(Wm[layer], wpk);
    gemm_wmma<<<cdiv(N / 16, 4), 128, 0, stream>>>(hbf, wpk, xw, N);

    // deg / norm / coef for this layer's edge weights
    fill_f32<<<cdiv(N, T), T, 0, stream>>>(deg, 1.f, N);
    fill_f32<<<cdiv(N, T), T, 0, stream>>>(score, 0.f, N);
    fill_f32<<<cdiv(NF, T), T, 0, stream>>>(agg, 0.f, NF);
    edge_deg<<<cdiv(E, T), T, 0, stream>>>(esrc, edst, maskCur, deg, E);
    calc_norm<<<cdiv(N, T), T, 0, stream>>>(deg, norm, N);
    calc_coef<<<cdiv(E, T), T, 0, stream>>>(esrc, edst, maskCur, norm, coef, E);

    // feature aggregation + combine + relu
    scatter_feat<<<cdiv((long long)E * 32, T), T, 0, stream>>>(esrc, edst, xw, coef, agg, E);
    combine_relu<<<cdiv(NF, T), T, 0, stream>>>(agg, xw, norm, bm[layer], hbuf, N);

    // score conv (shares deg/norm/coef)
    gemv_score<<<cdiv(N, 8), T, 0, stream>>>(hbuf, Wsm[layer], slin, N);
    scatter_score<<<cdiv(E, T), T, 0, stream>>>(esrc, edst, slin, coef, score, E);
    finalize_score<<<cdiv(N, T), T, 0, stream>>>(score, slin, norm, bsm[layer], N);

    // top-k pooling
    fill_f32<<<cdiv(G, T), T, 0, stream>>>(active, 0.f, G);
    accum_active<<<cdiv(N, T), T, 0, stream>>>(maskCur, batch, active, N);
    topk_mask<<<G, 256, 0, stream>>>(score, maskCur, starts, cnt, active, maskNext);
    apply_gate<<<cdiv(NF, T), T, 0, stream>>>(hbuf, score, maskNext, N);

    // readout accumulate
    readout<<<G, 128, 0, stream>>>(hbuf, maskNext, starts, cnt, z);

    curh = hbuf;
    float* tmp = maskCur; maskCur = maskNext; maskNext = tmp;
  }

  // MLP head
  linear_k<<<cdiv(G * 128, T), T, 0, stream>>>(z,  Wl1, bl1, z1, G, 256, 128, 1);
  linear_k<<<cdiv(G * 64,  T), T, 0, stream>>>(z1, Wl2, bl2, z2, G, 128, 64, 1);
  linear_k<<<cdiv(G * NCLS, T), T, 0, stream>>>(z2, Wl3, bl3, z3, G, 64, NCLS, 0);
  log_softmax_k<<<cdiv(G, T), T, 0, stream>>>(z3, (float*)d_out, G);
}
